// Adaptive_MSAB_90520730730929
// MI455X (gfx1250) — matrix operand order repaired and checked by reading, not compiled, not run on hardware
//
#include <hip/hip_runtime.h>
#include <math.h>

// ---------------- problem constants ----------------
#define Bn    8
#define Cc    64
#define Hh    128
#define Ww    128
#define NPIX  (Hh*Ww)        // 16384
#define BNr   (Bn*NPIX)      // 131072 token rows
#define HEADS 2
#define DHd   32

// ---------------- WMMA plumbing (gfx1250, wave32) ----------------
typedef __attribute__((ext_vector_type(16))) __bf16        v16bf;
typedef __attribute__((ext_vector_type(8)))  float         v8f;
typedef __attribute__((ext_vector_type(4)))  unsigned int  u32x4;

union FragBF { v16bf v; u32x4 q[2]; };   // 32 bytes: 16 bf16 lane elements

__device__ __forceinline__ unsigned short f2bfu(float x) {
  union { float f; unsigned u; } v; v.f = x;
  unsigned r = v.u + 0x7FFFu + ((v.u >> 16) & 1u);   // round-to-nearest-even
  return (unsigned short)(r >> 16);
}
__device__ __forceinline__ float gelu_f(float x) { return 0.5f*x*(1.0f + erff(x*0.70710678118654752f)); }
__device__ __forceinline__ float sigm_f(float x) { return 1.0f/(1.0f + expf(-x)); }

__device__ __forceinline__ v8f wmma_bf16(v16bf a, v16bf b, v8f c) {
  return __builtin_amdgcn_wmma_f32_16x16x32_bf16(false, a, false, b, (short)0, c, false, false);
}

// ---------------- LayerNorm: NCHW f32 input -> (B,N,C) bf16 row-major ----------------
__global__ void k_ln_nchw(const float* __restrict__ x, const float* __restrict__ g,
                          const float* __restrict__ bb, unsigned short* __restrict__ out) {
  int blk = blockIdx.x; int b = blk >> 9; int n0 = (blk & 511) << 5;   // N/32 = 512
  int nl = threadIdx.x & 31; int cg = threadIdx.x >> 5;                // 32 n x 8 c-groups
  const float* xb = x + (size_t)b*Cc*NPIX;
  float vals[8]; float s = 0.f, sq = 0.f;
#pragma unroll
  for (int i = 0; i < 8; ++i) {
    int c = cg*8 + i;
    float v = xb[(size_t)c*NPIX + n0 + nl];
    vals[i] = v; s += v; sq += v*v;
  }
  __shared__ float S[8][32], SQ[8][32], MU[32], RS[32];
  S[cg][nl] = s; SQ[cg][nl] = sq; __syncthreads();
  if (cg == 0) {
    float ts = 0.f, tq = 0.f;
    for (int j = 0; j < 8; ++j) { ts += S[j][nl]; tq += SQ[j][nl]; }
    float mu = ts*(1.0f/64.0f);
    MU[nl] = mu; RS[nl] = rsqrtf(tq*(1.0f/64.0f) - mu*mu + 1e-5f);
  }
  __syncthreads();
  float mu = MU[nl], rs = RS[nl];
#pragma unroll
  for (int i = 0; i < 8; ++i) {
    int c = cg*8 + i;
    out[((size_t)b*NPIX + n0 + nl)*Cc + c] = f2bfu((vals[i] - mu)*rs*g[c] + bb[c]);
  }
}

// ---------------- LayerNorm on x2 half of h1 (f32, row stride 128) -> f32 ----------------
__global__ void k_ln_rows(const float* __restrict__ h1, const float* __restrict__ g,
                          const float* __restrict__ bb, float* __restrict__ out) {
  int r = threadIdx.x >> 6; int c = threadIdx.x & 63;   // 4 rows x 64 channels
  size_t row = (size_t)blockIdx.x*4 + r;
  float v = h1[row*128 + 64 + c];
  __shared__ float buf[4][64]; __shared__ float mu_s[4], rs_s[4];
  buf[r][c] = v; __syncthreads();
  if (c == 0) {
    float s = 0.f, sq = 0.f;
    for (int j = 0; j < 64; ++j) { float t = buf[r][j]; s += t; sq += t*t; }
    float mu = s*(1.0f/64.0f);
    mu_s[r] = mu; rs_s[r] = rsqrtf(sq*(1.0f/64.0f) - mu*mu + 1e-5f);
  }
  __syncthreads();
  out[row*64 + c] = (v - mu_s[r])*rs_s[r]*g[c] + bb[c];
}

// ---------------- WMMA GEMM, fully compile-time specialized ----------------
// Y(M x ND) f32 = Xb(M x 64 bf16) @ W (+bias, opt GELU). Weights staged to LDS as bf16,
// n-major; TRANS resolved at stage time. Optional secondary bf16 output.
template<int ND, int TRANS, int ACT, bool HASBIAS, bool BF16OUT>
__global__ void k_gemm(const unsigned short* __restrict__ Xb, const float* __restrict__ Wm,
                       const float* __restrict__ bias, float* __restrict__ Y,
                       unsigned short* __restrict__ Ybf, int Mtiles, int ldw) {
  const int Kpad = 72;                                  // bf16 elems; rows 16B aligned
  __shared__ __align__(16) unsigned short WT[ND*Kpad];
#pragma unroll 4
  for (int i = threadIdx.x; i < ND*64; i += 256) {
    int n = i >> 6, k = i & 63;
    float wv = TRANS ? Wm[(size_t)n*ldw + k] : Wm[(size_t)k*ldw + n];
    WT[n*Kpad + k] = f2bfu(wv);
  }
  __syncthreads();
  int wave = (int)((blockIdx.x*blockDim.x + threadIdx.x) >> 5);
  if (wave >= Mtiles) return;                           // wave-uniform
  int lane = threadIdx.x & 31;
  int hl = lane >> 4, ml = lane & 15, nl = lane & 15;
  size_t m0 = (size_t)wave * 16;

  // A fragments for k=0..31 and k=32..63 (two contiguous 8-elem runs each)
  const u32x4* xr = (const u32x4*)(Xb + (m0 + ml)*64);
  FragBF a0, a1;
  a0.q[0] = xr[hl];     a0.q[1] = xr[2 + hl];
  a1.q[0] = xr[4 + hl]; a1.q[1] = xr[6 + hl];

#pragma unroll
  for (int nt = 0; nt < ND; nt += 16) {
    const u32x4* wrow = (const u32x4*)(WT + (nt + nl)*Kpad);
    FragBF b0, b1;
    b0.q[0] = wrow[hl];       b0.q[1] = wrow[2 + hl];
    b1.q[0] = wrow[4 + hl];   b1.q[1] = wrow[6 + hl];
    float bz = HASBIAS ? bias[nt + nl] : 0.0f;
    v8f acc;
#pragma unroll
    for (int r = 0; r < 8; ++r) acc[r] = bz;
    acc = wmma_bf16(a0.v, b0.v, acc);
    acc = wmma_bf16(a1.v, b1.v, acc);
#pragma unroll
    for (int r = 0; r < 8; ++r) {
      float v = acc[r];
      if (ACT == 1) v = gelu_f(v);
      size_t off = (m0 + r + 8*hl)*(size_t)ND + nt + nl;
      Y[off] = v;
      if (BF16OUT) Ybf[off] = f2bfu(v);
    }
  }
}

// ------- L2-normalize each (b,c) column over N; emit bf16 TRANSPOSED (B,C,N) -------------
__global__ void k_colnorm(const float* __restrict__ Qf, unsigned short* __restrict__ QT) {
  __shared__ float red[256]; __shared__ float inv_s;
  int bc = blockIdx.x; int b = bc >> 6, c = bc & 63; int tid = threadIdx.x;
  const float* base = Qf + (size_t)b*NPIX*Cc + c;
  float ss = 0.f;
  for (int n = tid; n < NPIX; n += 256) { float v = base[(size_t)n*Cc]; ss += v*v; }
  red[tid] = ss; __syncthreads();
  for (int off = 128; off > 0; off >>= 1) { if (tid < off) red[tid] += red[tid + off]; __syncthreads(); }
  if (tid == 0) inv_s = 1.0f/fmaxf(sqrtf(red[0]), 1e-12f);
  __syncthreads();
  float s = inv_s;
  unsigned short* orow = QT + (size_t)bc*NPIX;
  for (int n = tid; n < NPIX; n += 256) orow[n] = f2bfu(base[(size_t)n*Cc]*s);
}

// ------- attn[b,h,d,e] = sum_n k[d,n] q[e,n] * rescale[h]; bf16 (B,C,N) inputs, WMMA -----
__global__ void k_attn(const unsigned short* __restrict__ KT, const unsigned short* __restrict__ QT,
                       const float* __restrict__ rescale, float* __restrict__ att) {
  int wave = (int)((blockIdx.x*blockDim.x + threadIdx.x) >> 5);   // 64 waves total
  int lane = threadIdx.x & 31; int hl = lane >> 4, ml = lane & 15, nl = lane & 15;
  int b = wave >> 3, h = (wave >> 2) & 1, dt = (wave >> 1) & 1, et = wave & 1;
  const u32x4* arow = (const u32x4*)(KT + ((size_t)b*Cc + h*32 + dt*16 + ml)*NPIX);
  const u32x4* brow = (const u32x4*)(QT + ((size_t)b*Cc + h*32 + et*16 + nl)*NPIX);
  v8f acc = {0,0,0,0,0,0,0,0};
#pragma unroll 4
  for (int n0 = 0; n0 < NPIX; n0 += 32) {
    int qb = n0 >> 3;
    FragBF a, bm;
    a.q[0]  = arow[qb + hl];     a.q[1]  = arow[qb + 2 + hl];
    bm.q[0] = brow[qb + hl];     bm.q[1] = brow[qb + 2 + hl];
    acc = wmma_bf16(a.v, bm.v, acc);
  }
  float rs = rescale[h];
#pragma unroll
  for (int r = 0; r < 8; ++r) {
    int d = dt*16 + r + 8*hl; int e = et*16 + nl;
    att[(((size_t)b*HEADS + h)*DHd + d)*DHd + e] = acc[r]*rs;
  }
}

// ---------------- softmax over last dim (32); emits f32 + bf16 copies ----------------
__global__ void k_softmax(float* __restrict__ att, unsigned short* __restrict__ attbf) {
  int r = blockIdx.x*blockDim.x + threadIdx.x;
  if (r >= Bn*HEADS*DHd) return;
  float* row = att + (size_t)r*DHd;
  float mx = -3.4e38f;
  for (int e = 0; e < DHd; ++e) mx = fmaxf(mx, row[e]);
  float s = 0.f;
  for (int e = 0; e < DHd; ++e) { float v = expf(row[e] - mx); row[e] = v; s += v; }
  float inv = 1.0f/s;
  for (int e = 0; e < DHd; ++e) { float v = row[e]*inv; row[e] = v; attbf[(size_t)r*DHd + e] = f2bfu(v); }
}

// ------- O(B,N,C) bf16: o[n, h*32+d] = sum_e attn[b,h,d,e] * V[n, h*32+e]  (WMMA) --------
__global__ void k_oav(const unsigned short* __restrict__ attbf, const unsigned short* __restrict__ Vbf,
                      unsigned short* __restrict__ Obf) {
  int wave = (int)((blockIdx.x*blockDim.x + threadIdx.x) >> 5);
  if (wave >= BNr/16) return;
  int lane = threadIdx.x & 31; int hl = lane >> 4, ml = lane & 15, nl = lane & 15;
  size_t m0 = (size_t)wave*16;
  int b = (int)(m0 / NPIX);
  const u32x4* arow = (const u32x4*)(Vbf + (m0 + ml)*64);
#pragma unroll
  for (int h = 0; h < HEADS; ++h) {
    FragBF a;
    a.q[0] = arow[4*h + hl]; a.q[1] = arow[4*h + 2 + hl];
#pragma unroll
    for (int dt = 0; dt < 2; ++dt) {
      const u32x4* brow = (const u32x4*)(attbf + (((size_t)b*HEADS + h)*DHd + dt*16 + nl)*DHd);
      FragBF bm;
      bm.q[0] = brow[hl]; bm.q[1] = brow[2 + hl];
      v8f acc = {0,0,0,0,0,0,0,0};
      acc = wmma_bf16(a.v, bm.v, acc);
#pragma unroll
      for (int r = 0; r < 8; ++r)
        Obf[(m0 + r + 8*hl)*64 + h*32 + dt*16 + nl] = f2bfu(acc[r]);
    }
  }
}

// ---------------- depthwise 3x3 on v_out(NCHW view of f32 V) + BN + GELU -> convx NCHW ---
__global__ void k_dwconv_v(const float* __restrict__ Vb, const float* __restrict__ wdw,
                           const float* __restrict__ bdw, const float* __restrict__ g,
                           const float* __restrict__ bb, const float* __restrict__ mm,
                           const float* __restrict__ vv, float* __restrict__ convx) {
  size_t idx = (size_t)blockIdx.x*blockDim.x + threadIdx.x;
  if (idx >= (size_t)Bn*Cc*NPIX) return;
  int n = (int)(idx % NPIX); int bc = (int)(idx / NPIX); int c = bc & 63; int b = bc >> 6;
  int hh = n >> 7, ww = n & 127;
  float acc = bdw[c];
  for (int kh = 0; kh < 3; ++kh) {
    int y = hh + kh - 1; if (y < 0 || y >= Hh) continue;
    for (int kw = 0; kw < 3; ++kw) {
      int x = ww + kw - 1; if (x < 0 || x >= Ww) continue;
      acc += Vb[((size_t)b*NPIX + y*Ww + x)*Cc + c] * wdw[c*9 + kh*3 + kw];
    }
  }
  float sc = g[c]*rsqrtf(vv[c] + 1e-5f);
  convx[idx] = gelu_f((acc - mm[c])*sc + bb[c]);
}

// ---------------- mean+max pool over HW per (b,c) ----------------
__global__ void k_pool(const float* __restrict__ convx, float* __restrict__ pool) {
  __shared__ float ss[256], sm[256];
  int bc = blockIdx.x; int tid = threadIdx.x;
  const float* p = convx + (size_t)bc*NPIX;
  float s = 0.f, m = -3.4e38f;
  for (int i = tid; i < NPIX; i += 256) { float v = p[i]; s += v; m = fmaxf(m, v); }
  ss[tid] = s; sm[tid] = m; __syncthreads();
  for (int off = 128; off > 0; off >>= 1) {
    if (tid < off) { ss[tid] += ss[tid + off]; sm[tid] = fmaxf(sm[tid], sm[tid + off]); }
    __syncthreads();
  }
  if (tid == 0) {
    int b = bc >> 6, c = bc & 63;
    pool[b*2*Cc + c]      = ss[0]*(1.0f/(float)NPIX);
    pool[b*2*Cc + Cc + c] = sm[0];
  }
}

// ---------------- channel attention: 2C -> 8 (gelu) -> C (sigmoid) ----------------
__global__ void k_ci(const float* __restrict__ pool, const float* __restrict__ w1,
                     const float* __restrict__ b1, const float* __restrict__ w2,
                     const float* __restrict__ b2, float* __restrict__ ci) {
  __shared__ float t8[8];
  int b = blockIdx.x; int c = threadIdx.x;
  if (c < 8) {
    float a = b1[c];
    for (int i = 0; i < 2*Cc; ++i) a += pool[b*2*Cc + i]*w1[c*2*Cc + i];
    t8[c] = gelu_f(a);
  }
  __syncthreads();
  float a = b2[c];
#pragma unroll
  for (int j = 0; j < 8; ++j) a += t8[j]*w2[c*8 + j];
  ci[b*Cc + c] = sigm_f(a);
}

// ---------------- spatial attention chain ----------------
__global__ void k_si1(const float* __restrict__ convx, const float* __restrict__ w,
                      const float* __restrict__ bb, float* __restrict__ si1) {
  size_t idx = (size_t)blockIdx.x*blockDim.x + threadIdx.x;
  if (idx >= (size_t)Bn*4*NPIX) return;
  int n = (int)(idx % NPIX); int bj = (int)(idx / NPIX); int j = bj & 3; int b = bj >> 2;
  float a = bb[j];
  for (int c = 0; c < Cc; ++c) a += convx[((size_t)b*Cc + c)*NPIX + n] * w[j*Cc + c];
  si1[idx] = a;
}

__global__ void k_si2(const float* __restrict__ si1, const float* __restrict__ w,
                      const float* __restrict__ bb, const float* __restrict__ g,
                      const float* __restrict__ beta, const float* __restrict__ mm,
                      const float* __restrict__ vv, float* __restrict__ si2) {
  size_t idx = (size_t)blockIdx.x*blockDim.x + threadIdx.x;
  if (idx >= (size_t)Bn*4*NPIX) return;
  int n = (int)(idx % NPIX); int bj = (int)(idx / NPIX); int j = bj & 3;
  int hh = n >> 7, ww = n & 127;
  float acc = bb[j];
  for (int kh = 0; kh < 3; ++kh) {
    int y = hh + kh - 1; if (y < 0 || y >= Hh) continue;
    for (int kw = 0; kw < 3; ++kw) {
      int x = ww + kw - 1; if (x < 0 || x >= Ww) continue;
      acc += si1[(size_t)bj*NPIX + y*Ww + x] * w[j*9 + kh*3 + kw];
    }
  }
  float sc = g[j]*rsqrtf(vv[j] + 1e-5f);
  si2[idx] = gelu_f((acc - mm[j])*sc + beta[j]);
}

__global__ void k_si3(const float* __restrict__ si2, const float* __restrict__ w3,
                      const float* __restrict__ b3, float* __restrict__ si) {
  size_t idx = (size_t)blockIdx.x*blockDim.x + threadIdx.x;
  if (idx >= (size_t)Bn*NPIX) return;
  int n = (int)(idx % NPIX); int b = (int)(idx / NPIX);
  int hh = n >> 7, ww = n & 127;
  float acc = b3[0];
  for (int j = 0; j < 4; ++j)
    for (int kh = 0; kh < 3; ++kh) {
      int y = hh + kh - 1; if (y < 0 || y >= Hh) continue;
      for (int kw = 0; kw < 3; ++kw) {
        int x = ww + kw - 1; if (x < 0 || x >= Ww) continue;
        acc += si2[((size_t)(b*4 + j))*NPIX + y*Ww + x] * w3[j*9 + kh*3 + kw];
      }
    }
  si[idx] = sigm_f(acc);
}

// ---------------- mix = convx*ci + attnx*si -> bf16 rows (B,N,C) ----------------
__global__ void k_mix(const float* __restrict__ convx, const float* __restrict__ ci,
                      const float* __restrict__ at, const float* __restrict__ si,
                      unsigned short* __restrict__ mixb) {
  size_t idx = (size_t)blockIdx.x*blockDim.x + threadIdx.x;
  if (idx >= (size_t)BNr*Cc) return;
  int c = (int)(idx & 63); size_t bnv = idx >> 6;
  int n = (int)(bnv % NPIX); int b = (int)(bnv / NPIX);
  float v = convx[((size_t)b*Cc + c)*NPIX + n]*ci[b*Cc + c] + at[idx]*si[bnv];
  mixb[idx] = f2bfu(v);
}

// ---------------- residual: out(NCHW) = x_in + transpose(yo) ----------------
__global__ void k_resid(const float* __restrict__ x_in, const float* __restrict__ yo,
                        float* __restrict__ outb) {
  size_t idx = (size_t)blockIdx.x*blockDim.x + threadIdx.x;
  if (idx >= (size_t)Bn*Cc*NPIX) return;
  int n = (int)(idx % NPIX); int bc = (int)(idx / NPIX); int c = bc & 63; int b = bc >> 6;
  outb[idx] = x_in[idx] + yo[((size_t)b*NPIX + n)*Cc + c];
}

// ---------------- spatial gate depthwise conv on x2 (B,N,C) f32 -> f32 ----------------
__global__ void k_sg(const float* __restrict__ x2, const float* __restrict__ wsg,
                     const float* __restrict__ bsg, float* __restrict__ x2c) {
  size_t idx = (size_t)blockIdx.x*blockDim.x + threadIdx.x;
  if (idx >= (size_t)BNr*Cc) return;
  int c = (int)(idx & 63); size_t bnv = idx >> 6;
  int n = (int)(bnv % NPIX); int b = (int)(bnv / NPIX);
  int hh = n >> 7, ww = n & 127;
  float acc = bsg[c];
  for (int kh = 0; kh < 3; ++kh) {
    int y = hh + kh - 1; if (y < 0 || y >= Hh) continue;
    for (int kw = 0; kw < 3; ++kw) {
      int x = ww + kw - 1; if (x < 0 || x >= Ww) continue;
      acc += x2[((size_t)b*NPIX + y*Ww + x)*Cc + c] * wsg[c*9 + kh*3 + kw];
    }
  }
  x2c[idx] = acc;
}

// ---------------- prod = x1 * x2conv -> bf16 ----------------
__global__ void k_prod(const float* __restrict__ h1, const float* __restrict__ x2c,
                       unsigned short* __restrict__ prodb) {
  size_t idx = (size_t)blockIdx.x*blockDim.x + threadIdx.x;
  if (idx >= (size_t)BNr*Cc) return;
  int c = (int)(idx & 63); size_t bnv = idx >> 6;
  prodb[idx] = f2bfu(h1[bnv*128 + c] * x2c[idx]);
}

// ---------------- final: d_out(NCHW) = outb + transpose(ffo) ----------------
__global__ void k_final(const float* __restrict__ outb, const float* __restrict__ ffo,
                        float* __restrict__ dout) {
  size_t idx = (size_t)blockIdx.x*blockDim.x + threadIdx.x;
  if (idx >= (size_t)Bn*Cc*NPIX) return;
  int n = (int)(idx % NPIX); int bc = (int)(idx / NPIX); int c = bc & 63; int b = bc >> 6;
  dout[idx] = outb[idx] + ffo[((size_t)b*NPIX + n)*Cc + c];
}

// =========================================================================================
extern "C" void kernel_launch(void* const* d_in, const int* in_sizes, int n_in,
                              void* d_out, int out_size, void* d_ws, size_t ws_size,
                              hipStream_t stream) {
  (void)in_sizes; (void)n_in; (void)out_size; (void)ws_size;
  const float* x_in  = (const float*)d_in[0];
  const float* g1    = (const float*)d_in[1];
  const float* b1    = (const float*)d_in[2];
  const float* wq    = (const float*)d_in[3];
  const float* wk    = (const float*)d_in[4];
  const float* wv    = (const float*)d_in[5];
  const float* resc  = (const float*)d_in[6];
  const float* wproj = (const float*)d_in[7];
  const float* bproj = (const float*)d_in[8];
  const float* w_dw  = (const float*)d_in[9];
  const float* b_dw  = (const float*)d_in[10];
  const float* bn1_g = (const float*)d_in[11];
  const float* bn1_b = (const float*)d_in[12];
  const float* bn1_m = (const float*)d_in[13];
  const float* bn1_v = (const float*)d_in[14];
  const float* w_ci1 = (const float*)d_in[15];
  const float* b_ci1 = (const float*)d_in[16];
  const float* w_ci2 = (const float*)d_in[17];
  const float* b_ci2 = (const float*)d_in[18];
  const float* w_si1 = (const float*)d_in[19];
  const float* b_si1 = (const float*)d_in[20];
  const float* w_si2 = (const float*)d_in[21];
  const float* b_si2 = (const float*)d_in[22];
  const float* bn2_g = (const float*)d_in[23];
  const float* bn2_b = (const float*)d_in[24];
  const float* bn2_m = (const float*)d_in[25];
  const float* bn2_v = (const float*)d_in[26];
  const float* w_si3 = (const float*)d_in[27];
  const float* b_si3 = (const float*)d_in[28];
  const float* w_out = (const float*)d_in[29];
  const float* g2    = (const float*)d_in[30];
  const float* b2    = (const float*)d_in[31];
  const float* w_fc1 = (const float*)d_in[32];
  const float* b_fc1 = (const float*)d_in[33];
  const float* sg_g  = (const float*)d_in[34];
  const float* sg_b  = (const float*)d_in[35];
  const float* w_sg  = (const float*)d_in[36];
  const float* b_sg  = (const float*)d_in[37];
  const float* w_fc2 = (const float*)d_in[38];
  const float* b_fc2 = (const float*)d_in[39];
  float* dout = (float*)d_out;

  float* ws = (float*)d_ws;
  const size_t BNC = (size_t)BNr*Cc;       // 8,388,608 elements
  // ---- f32 slots (aliased by lifetime) ----
  float* S0 = ws + 0*BNC;   // Q -> YO -> X2C -> FFO
  float* S1 = ws + 1*BNC;   // K -> CONVX -> H1[0:BNC]
  float* S2 = ws + 2*BNC;   // V -> H1[BNC:2BNC]
  float* S3 = ws + 3*BNC;   // AT -> X2
  float* S4 = ws + 4*BNC;   // OUTB
  // ---- bf16 slots (each BNC ushorts = BNC/2 floats) ----
  unsigned short* Hb = (unsigned short*)(ws + 5*BNC);
  unsigned short* H0 = Hb + 0*BNC;   // XSb -> FFb
  unsigned short* H1b= Hb + 1*BNC;   // QT -> Obf -> MIXb -> PRODb
  unsigned short* H2b= Hb + 2*BNC;   // KT
  unsigned short* H3b= Hb + 3*BNC;   // Vbf
  // ---- aux region ----
  float* AUX = ws + 5*BNC + 2*BNC;   // after 4 bf16 slots (= 2*BNC floats)
  float* ATT  = AUX;                             // 16384
  float* POOL = ATT + (size_t)Bn*HEADS*DHd*DHd;  // 1024
  float* CI   = POOL + (size_t)Bn*2*Cc;          // 512
  float* SI1  = CI + (size_t)Bn*Cc;              // B*4*N
  float* SI2  = SI1 + (size_t)Bn*4*NPIX;
  float* SI   = SI2 + (size_t)Bn*4*NPIX;         // B*N
  unsigned short* ATTBF = (unsigned short*)(SI + (size_t)Bn*NPIX);  // 16384 ushorts

  float* Q = S0; float* K = S1; float* V = S2;
  float* AT = S3; float* CONVX = S1; float* YO = S0; float* OUTB = S4;
  float* H1 = S1; /* spans S1..S2 (2*BNC) */
  float* X2 = S3; float* X2C = S0; float* FFO = S0;
  unsigned short* XSb = H0; unsigned short* FFb = H0;
  unsigned short* QT = H1b; unsigned short* Obf = H1b;
  unsigned short* MIXb = H1b; unsigned short* PRODb = H1b;
  unsigned short* KT = H2b; unsigned short* Vbf = H3b;

  const int MT = BNr/16;                 // 8192 M-tiles
  const int GEMM_BLOCKS = MT/8;          // 1024 blocks of 8 waves
  const int EL_BLOCKS = (int)(BNC/256);  // 32768

  // 1) LN1 -> bf16 rows
  k_ln_nchw<<<Bn*(NPIX/32), 256, 0, stream>>>(x_in, g1, b1, XSb);
  // 2-4) QKV (V also emits bf16 copy for attn*V)
  k_gemm<64,0,0,false,false><<<GEMM_BLOCKS, 256, 0, stream>>>(XSb, wq, nullptr, Q, nullptr, MT, 64);
  k_gemm<64,0,0,false,false><<<GEMM_BLOCKS, 256, 0, stream>>>(XSb, wk, nullptr, K, nullptr, MT, 64);
  k_gemm<64,0,0,false,true ><<<GEMM_BLOCKS, 256, 0, stream>>>(XSb, wv, nullptr, V, Vbf, MT, 64);
  // 5) L2 normalize q,k per (b,channel) over N -> bf16 transposed (B,C,N)
  k_colnorm<<<Bn*Cc, 256, 0, stream>>>(Q, QT);
  k_colnorm<<<Bn*Cc, 256, 0, stream>>>(K, KT);
  // 6-7) attention matrix + softmax (emits bf16 copy)
  k_attn<<<16, 128, 0, stream>>>(KT, QT, resc, ATT);
  k_softmax<<<(Bn*HEADS*DHd + 255)/256, 256, 0, stream>>>(ATT, ATTBF);
  // 8) o = attn @ v -> bf16 rows
  k_oav<<<GEMM_BLOCKS, 256, 0, stream>>>(ATTBF, Vbf, Obf);
  // 9) attnx = o @ wproj + bproj (f32 rows)
  k_gemm<64,0,0,true,false><<<GEMM_BLOCKS, 256, 0, stream>>>(Obf, wproj, bproj, AT, nullptr, MT, 64);
  // 10) convx = gelu(bn(dwconv(v_out)))
  k_dwconv_v<<<EL_BLOCKS, 256, 0, stream>>>(V, w_dw, b_dw, bn1_g, bn1_b, bn1_m, bn1_v, CONVX);
  // 11-12) channel attention
  k_pool<<<Bn*Cc, 256, 0, stream>>>(CONVX, POOL);
  k_ci<<<Bn, 64, 0, stream>>>(POOL, w_ci1, b_ci1, w_ci2, b_ci2, CI);
  // 13-15) spatial attention
  k_si1<<<(int)((size_t)Bn*4*NPIX/256), 256, 0, stream>>>(CONVX, w_si1, b_si1, SI1);
  k_si2<<<(int)((size_t)Bn*4*NPIX/256), 256, 0, stream>>>(SI1, w_si2, b_si2, bn2_g, bn2_b, bn2_m, bn2_v, SI2);
  k_si3<<<(int)((size_t)Bn*NPIX/256), 256, 0, stream>>>(SI2, w_si3, b_si3, SI);
  // 16) mix -> bf16 rows
  k_mix<<<EL_BLOCKS, 256, 0, stream>>>(CONVX, CI, AT, SI, MIXb);
  // 17) 1x1 conv w_out (transposed weight layout)
  k_gemm<64,1,0,false,false><<<GEMM_BLOCKS, 256, 0, stream>>>(MIXb, w_out, nullptr, YO, nullptr, MT, 64);
  // 18) residual into NCHW
  k_resid<<<EL_BLOCKS, 256, 0, stream>>>(x_in, YO, OUTB);
  // 19) LN2 -> bf16 rows
  k_ln_nchw<<<Bn*(NPIX/32), 256, 0, stream>>>(OUTB, g2, b2, FFb);
  // 20) fc1 + GELU fused (f32 out, Nd=128)
  k_gemm<128,0,1,true,false><<<GEMM_BLOCKS, 256, 0, stream>>>(FFb, w_fc1, b_fc1, H1, nullptr, MT, 128);
  // 21) LN of x2 half
  k_ln_rows<<<BNr/4, 256, 0, stream>>>(H1, sg_g, sg_b, X2);
  // 22) depthwise spatial gate conv
  k_sg<<<EL_BLOCKS, 256, 0, stream>>>(X2, w_sg, b_sg, X2C);
  // 23) x1 * x2 -> bf16
  k_prod<<<EL_BLOCKS, 256, 0, stream>>>(H1, X2C, PRODb);
  // 24) fc2
  k_gemm<64,0,0,true,false><<<GEMM_BLOCKS, 256, 0, stream>>>(PRODb, w_fc2, b_fc2, FFO, MT ? nullptr : nullptr, MT, 64);
  // 25) final residual -> d_out (NCHW)
  k_final<<<EL_BLOCKS, 256, 0, stream>>>(OUTB, FFO, dout);
}
